// TorchHeteroMPNN_80719615361489
// MI455X (gfx1250) — matrix operand order repaired and checked by reading, not compiled, not run on hardware
//
#include <hip/hip_runtime.h>

// ---------------------------------------------------------------------------
// CDNA5 (gfx1250) hetero-MPNN. wave32 WMMA f32_16x16x32_f16 for every linear.
// All per-edge linears are commuted to per-node linears (gather commutes with
// right-matmul), so edges become pure gather + atomic scatter-add.
// LDS A tile uses a padded row stride (K+8 halfs) so the per-lane
// ds_load_b128 fragment fetches are LDS bank-conflict-free.
// ---------------------------------------------------------------------------

typedef _Float16 half_t;
typedef __attribute__((ext_vector_type(16))) _Float16 v16h;
typedef __attribute__((ext_vector_type(8)))  _Float16 v8h;
typedef __attribute__((ext_vector_type(4)))  _Float16 v4h;
typedef __attribute__((ext_vector_type(8)))  float    v8f;

#define HD   128   // hidden dim
#define APAD 8     // LDS row padding in halfs (16B): row stride K+8 halfs
                   // -> row stride (K+8)/2 dwords -> rows start 4 banks apart

// ---------------------------------------------------------------------------
// Pre-swizzle a fp32 weight [K][128] into f16 WMMA B-fragment order.
// Fragment layout (CDNA5 ISA 16-bit B 32x16): for n-strip s (8 strips of 16
// cols), k-chunk c (K/32 chunks), lane l: halfs j=0..15 hold
// B[k = c*32 + (l>>4)*8 + j + (j>>3)*8][n = s*16 + (l&15)], contiguous per lane so the
// GEMM loads each fragment as two 16-byte global loads.
// ---------------------------------------------------------------------------
__global__ __launch_bounds__(256)
void prep_weight_kernel(const float* __restrict__ W, half_t* __restrict__ out, int K) {
    int t = blockIdx.x * blockDim.x + threadIdx.x;
    int KC = K >> 5;
    int total = 8 * KC * 32;
    if (t >= total) return;
    int lane  = t & 31;
    int c     = (t >> 5) % KC;
    int s     = t / (32 * KC);
    int kbase = c * 32 + ((lane >> 4) << 3);
    int n     = (s << 4) + (lane & 15);
    half_t* o = out + (size_t)t * 16;
#pragma unroll
    for (int j = 0; j < 16; ++j)
        o[j] = (half_t)W[(size_t)(kbase + j + ((j >> 3) << 3)) * HD + n];
}

// ---------------------------------------------------------------------------
// Generic GEMM: out[M,128] = relu(A[M,K] @ W[K,128] + bias), f32 accum.
//   mode 0: A = A16 (f16 hidden state)
//   mode 1: A = A32 (fp32 raw features)            [projection]
//   mode 2: A = f16(A16 + A32) (h + agg)           [update; in-place safe]
// Block = 256 threads = 8 waves; block owns 16 rows; wave w owns cols
// [16w, 16w+16). A tile staged once in LDS (padded stride), shared by all
// 8 waves.
// ---------------------------------------------------------------------------
__global__ __launch_bounds__(256)
void gemm16_relu_kernel(int M, int K, int mode,
                        const half_t* __restrict__ A16,
                        const float*  __restrict__ A32,
                        const half_t* __restrict__ Wfrag,
                        const float*  __restrict__ bias,
                        half_t* __restrict__ out) {
    __shared__ __align__(16) half_t As[16 * (HD + APAD)];

    const int tid = threadIdx.x;
    const int m0  = blockIdx.x << 4;
    const int ldA = K + APAD;          // padded LDS row stride (halfs)

    // --- stage A tile (16 x K) into LDS (zeros past M) ---
    const int total = 16 * K;
    for (int i = tid; i < total; i += 256) {
        int r = i / K;
        int c = i - r * K;
        int m = m0 + r;
        float v = 0.0f;
        if (m < M) {
            size_t off = (size_t)m * K + c;
            if (mode == 0)      v = (float)A16[off];
            else if (mode == 1) v = A32[off];
            else                v = (float)A16[off] + A32[off];
        }
        As[r * ldA + c] = (half_t)v;
    }
    __syncthreads();

    const int wid  = tid >> 5;
    const int lane = tid & 31;
    const int lh   = lane >> 4;   // 0 | 1 (lane half)
    const int l15  = lane & 15;
    const int KC   = K >> 5;

    v8f acc = {};
    // A fragment (16-bit 16x32 layout): lane l, row m=l&15,
    //   halfs 0..7  -> k = c*32 + (l>>4)*8 + j
    //   halfs 8..15 -> k = c*32 + 16 + (l>>4)*8 + (j-8)
    const half_t* arow = &As[l15 * ldA + (lh << 3)];
    const half_t* bptr = Wfrag + ((size_t)(wid * KC) * 32 + lane) * 16;

    for (int c = 0; c < KC; ++c) {
        union { v16h v; v8h h[2]; } a, b;
        const half_t* ap = arow + (c << 5);
        a.h[0] = *(const v8h*)(ap);        // ds_load_b128 (conflict-free)
        a.h[1] = *(const v8h*)(ap + 16);   // ds_load_b128 (conflict-free)
        const half_t* bp = bptr + (size_t)c * (32 * 16);
        b.h[0] = *(const v8h*)(bp);        // global_load_b128
        b.h[1] = *(const v8h*)(bp + 8);    // global_load_b128
        acc = __builtin_amdgcn_wmma_f32_16x16x32_f16(
                  false, a.v, false, b.v, (short)0, acc, false, false);
    }

    // --- epilogue: bias + relu, write f16 ---
    const int n  = (wid << 4) + l15;
    const float bn = bias[n];
#pragma unroll
    for (int r = 0; r < 8; ++r) {
        int m = m0 + r + (lh << 3);   // D layout: VGPR r -> M = r + 8*(lane>>4)
        if (m < M) {
            float v = acc[r] + bn;
            out[(size_t)m * HD + n] = (half_t)(v > 0.0f ? v : 0.0f);
        }
    }
}

// ---------------------------------------------------------------------------
// Edge scatter: one wave per edge. Coalesced 256B f16 gather of msg[src],
// 128 f32 atomic adds into agg[dst] (4 per lane).
// ---------------------------------------------------------------------------
__global__ __launch_bounds__(256)
void edge_scatter_kernel(const half_t* __restrict__ msg,
                         const int* __restrict__ src,
                         const int* __restrict__ dst,
                         float* __restrict__ agg, int E) {
    int gid = blockIdx.x * blockDim.x + threadIdx.x;
    int e = gid >> 5;
    if (e >= E) return;
    int l = gid & 31;
    int s = src[e], d = dst[e];
    const v4h m4 = *(const v4h*)(msg + (size_t)s * HD + (l << 2));
    float* ap = agg + (size_t)d * HD + (l << 2);
    atomicAdd(ap + 0, (float)m4[0]);
    atomicAdd(ap + 1, (float)m4[1]);
    atomicAdd(ap + 2, (float)m4[2]);
    atomicAdd(ap + 3, (float)m4[3]);
}

// ---------------------------------------------------------------------------
// Head tail: out[u,0:2] = t1[u,:] @ Wh2[128,2] + bh2. Tiny; one thread/user.
// ---------------------------------------------------------------------------
__global__ __launch_bounds__(256)
void head2_kernel(const half_t* __restrict__ t1, const float* __restrict__ Wh2,
                  const float* __restrict__ bh2, float* __restrict__ out, int NU) {
    int u = blockIdx.x * blockDim.x + threadIdx.x;
    if (u >= NU) return;
    float a0 = bh2[0], a1 = bh2[1];
    const half_t* row = t1 + (size_t)u * HD;
#pragma unroll 8
    for (int k = 0; k < HD; ++k) {
        float v = (float)row[k];
        a0 += v * Wh2[2 * k];
        a1 += v * Wh2[2 * k + 1];
    }
    out[2 * u]     = a0;
    out[2 * u + 1] = a1;
}

// ---------------------------------------------------------------------------
extern "C" void kernel_launch(void* const* d_in, const int* in_sizes, int n_in,
                              void* d_out, int out_size, void* d_ws, size_t ws_size,
                              hipStream_t stream) {
    const int NU = in_sizes[0] / 64;
    const int ND = in_sizes[1] / 32;
    const int NI = in_sizes[2] / 32;
    const int NP = in_sizes[3] / 32;
    const int E  = in_sizes[36];

    const float* x_user   = (const float*)d_in[0];
    const float* x_device = (const float*)d_in[1];
    const float* x_ip     = (const float*)d_in[2];
    const float* x_phone  = (const float*)d_in[3];

    // biases (fp32, used directly)
    const float* bp_user   = (const float*)d_in[5];
    const float* bu_user   = (const float*)d_in[7];
    const float* bp_device = (const float*)d_in[9];
    const float* bu_device = (const float*)d_in[11];
    const float* bp_ip     = (const float*)d_in[13];
    const float* bu_ip     = (const float*)d_in[15];
    const float* bp_phone  = (const float*)d_in[17];
    const float* bu_phone  = (const float*)d_in[19];
    const float* br_rel[6] = { (const float*)d_in[21], (const float*)d_in[23],
                               (const float*)d_in[25], (const float*)d_in[27],
                               (const float*)d_in[29], (const float*)d_in[31] };
    const float* bh1 = (const float*)d_in[33];
    const float* Wh2 = (const float*)d_in[34];
    const float* bh2 = (const float*)d_in[35];

    // ---- workspace carve ----
    char* p = (char*)d_ws;
    auto wsalloc = [&](size_t bytes) -> void* {
        void* r = (void*)p;
        p += (bytes + 255) & ~(size_t)255;
        return r;
    };
    half_t* h_user   = (half_t*)wsalloc((size_t)NU * HD * sizeof(half_t));
    half_t* h_device = (half_t*)wsalloc((size_t)ND * HD * sizeof(half_t));
    half_t* h_ip     = (half_t*)wsalloc((size_t)NI * HD * sizeof(half_t));
    half_t* h_phone  = (half_t*)wsalloc((size_t)NP * HD * sizeof(half_t));

    const size_t aggElems = (size_t)(NU + ND + NI + NP) * HD;
    float* agg_base   = (float*)wsalloc(aggElems * sizeof(float));
    float* agg_user   = agg_base;
    float* agg_device = agg_user   + (size_t)NU * HD;
    float* agg_ip     = agg_device + (size_t)ND * HD;
    float* agg_phone  = agg_ip     + (size_t)NI * HD;

    half_t* msg = (half_t*)wsalloc((size_t)NU * HD * sizeof(half_t)); // max-src scratch

    // ---- weight fragment prep (fp32 -> f16 fragment order) ----
    // order: Wp_user, Wp_device, Wp_ip, Wp_phone, Wu_user..Wu_phone,
    //        Wr_ud, Wr_du, Wr_ui, Wr_iu, Wr_up, Wr_pu, Wh1
    const int widx[15] = { 4, 8, 12, 16,  6, 10, 14, 18,
                           20, 22, 24, 26, 28, 30,  32 };
    const int wK[15]   = { 64, 32, 32, 32, 128, 128, 128, 128,
                           128, 128, 128, 128, 128, 128, 128 };
    half_t* wf[15];
    for (int i = 0; i < 15; ++i) {
        wf[i] = (half_t*)wsalloc((size_t)wK[i] * HD * sizeof(half_t));
        int threads = 8 * (wK[i] >> 5) * 32;
        prep_weight_kernel<<<(threads + 255) / 256, 256, 0, stream>>>(
            (const float*)d_in[widx[i]], wf[i], wK[i]);
    }

    auto gemm = [&](int M, int K, int mode, const half_t* A16, const float* A32,
                    const half_t* W, const float* b, half_t* out) {
        gemm16_relu_kernel<<<(M + 15) / 16, 256, 0, stream>>>(M, K, mode, A16, A32, W, b, out);
    };

    // ---- input projections: h_t = relu(x_t @ Wp_t + bp_t) ----
    gemm(NU, 64, 1, nullptr, x_user,   wf[0], bp_user,   h_user);
    gemm(ND, 32, 1, nullptr, x_device, wf[1], bp_device, h_device);
    gemm(NI, 32, 1, nullptr, x_ip,     wf[2], bp_ip,     h_ip);
    gemm(NP, 32, 1, nullptr, x_phone,  wf[3], bp_phone,  h_phone);

    // relation tables: src hidden, src count, frag idx, edges, dst agg
    const half_t* r_hsrc[6] = { h_user, h_device, h_user, h_ip, h_user, h_phone };
    const int     r_msrc[6] = { NU, ND, NU, NI, NU, NP };
    const int     r_wf[6]   = { 8, 9, 10, 11, 12, 13 };
    const int*    r_es[6]   = { (const int*)d_in[36], (const int*)d_in[38],
                                (const int*)d_in[40], (const int*)d_in[42],
                                (const int*)d_in[44], (const int*)d_in[46] };
    const int*    r_ed[6]   = { (const int*)d_in[37], (const int*)d_in[39],
                                (const int*)d_in[41], (const int*)d_in[43],
                                (const int*)d_in[45], (const int*)d_in[47] };
    float*        r_agg[6]  = { agg_device, agg_user, agg_ip, agg_user, agg_phone, agg_user };

    const int escatterBlocks = (E * 32 + 255) / 256;

    // ---- 2 message-passing rounds (shared weights) ----
    for (int round = 0; round < 2; ++round) {
        hipMemsetAsync(agg_base, 0, aggElems * sizeof(float), stream);
        for (int r = 0; r < 6; ++r) {
            // per-node relation linear (commuted): msg = relu(h_src @ Wr + br)
            gemm(r_msrc[r], 128, 0, r_hsrc[r], nullptr, wf[r_wf[r]], br_rel[r], msg);
            // edges: agg[dst] += msg[src]
            edge_scatter_kernel<<<escatterBlocks, 256, 0, stream>>>(
                msg, r_es[r], r_ed[r], r_agg[r], E);
        }
        // updates: h_t = relu((h_t + agg_t) @ Wu_t + bu_t)   (in-place safe)
        gemm(NU, 128, 2, h_user,   agg_user,   wf[4], bu_user,   h_user);
        gemm(ND, 128, 2, h_device, agg_device, wf[5], bu_device, h_device);
        gemm(NI, 128, 2, h_ip,     agg_ip,     wf[6], bu_ip,     h_ip);
        gemm(NP, 128, 2, h_phone,  agg_phone,  wf[7], bu_phone,  h_phone);
    }

    // ---- user head ----
    gemm(NU, 128, 0, h_user, nullptr, wf[14], bh1, msg);  // t1 = relu(h_user@Wh1+bh1)
    head2_kernel<<<(NU + 255) / 256, 256, 0, stream>>>(msg, Wh2, bh2, (float*)d_out, NU);
}
